// MultiHeadAlibiSelfAttentionLayer_34540126994482
// MI455X (gfx1250) — matrix operand order repaired and checked by reading, compiled, not run on hardware
//
#include <hip/hip_runtime.h>

// ---------------------------------------------------------------------------
// MultiHeadAlibiSelfAttentionLayer for MI455X (gfx1250, wave32, WMMA bf16)
// B=4, S=2048, D_IN=1024, HID=1024, H=16, DH=64
// GEMMs + flash attention all on v_wmma_f32_16x16x32_bf16; GEMM tiles staged
// with global_load_async_to_lds_b128 (ASYNCcnt) and double-buffered LDS.
// ---------------------------------------------------------------------------

typedef __bf16 bf16;
typedef __attribute__((ext_vector_type(16))) bf16  v16bf;
typedef __attribute__((ext_vector_type(8)))  bf16  v8bf;
typedef __attribute__((ext_vector_type(8)))  float v8f;

#define B_   4
#define S_   2048
#define DIN  1024
#define HID_ 1024
#define H_   16
#define DH_  64
#define M_   (B_ * S_)   // 8192 rows

// ---- WMMA helpers ---------------------------------------------------------

__device__ __forceinline__ v16bf pack2(v8bf lo, v8bf hi) {
  v16bf r;
#pragma unroll
  for (int i = 0; i < 8; ++i) { r[i] = lo[i]; r[i + 8] = hi[i]; }
  return r;
}

__device__ __forceinline__ v8f wmma_bf16(v16bf a, v16bf b, v8f c) {
  // D = A(16x32) * B(32x16) + C(16x16), f32 accumulate
  return __builtin_amdgcn_wmma_f32_16x16x32_bf16(
      /*neg_a=*/false, a, /*neg_b=*/false, b,
      /*c_mod=*/(short)0, c, /*reuse_a=*/false, /*reuse_b=*/false);
}

// Async memory->LDS copy, 16 bytes per lane, tracked by ASYNCcnt.
__device__ __forceinline__ void async_ld16(unsigned lds_off, const bf16* g) {
  asm volatile("global_load_async_to_lds_b128 %0, %1, off"
               :: "v"(lds_off), "v"(g) : "memory");
}

// ---- fp32 -> bf16 conversion kernels --------------------------------------

__global__ void mha_cvt_bf16(const float* __restrict__ in,
                             bf16* __restrict__ out, int n) {
  int i = blockIdx.x * 256 + threadIdx.x;
  if (i < n) out[i] = (bf16)in[i];
}

// W is [K][N] row-major fp32; produce Wt [N][K] bf16 so WMMA B-fragments
// (lane = column, K contiguous per lane) are contiguous loads.
__global__ void mha_cvt_bf16_T(const float* __restrict__ in,
                               bf16* __restrict__ out, int K, int N) {
  int i = blockIdx.x * 256 + threadIdx.x;
  if (i < K * N) {
    int k = i / N, n = i % N;
    out[n * K + k] = (bf16)in[i];
  }
}

// ---- Tiled bf16 GEMM: C[M,N] = A[M,K] * W[K,N] + bias ---------------------
// A: bf16 [M][K] row-major, Wt: bf16 [N][K] (pre-transposed).
// mode 0: fp32 out [M][N]                (final projection -> d_out)
// mode 1: bf16 out [B][H][S][DH]         (Q, K for attention)
// mode 2: bf16 out [B][H][DH][S]         (V transposed for attention)
// Block: 256 threads (8 waves), tile 128x128, K-step 32, double-buffered LDS
// filled by global_load_async_to_lds_b128.
// Wave grid 4x2: each wave computes 32 rows x 64 cols = 2x4 WMMA tiles.

#define TM 128
#define TN 128
#define TK 32
#define NT (DIN / TK)   // 32 K-tiles
#define LSTR 40         // LDS row stride in bf16 (80B = 20 banks, 16B aligned)
#define TILE_BYTES (TM * LSTR * 2)

__global__ __launch_bounds__(256)
void mha_gemm_bf16(const bf16* __restrict__ A, const bf16* __restrict__ Wt,
                   const float* __restrict__ bias, float* __restrict__ outF,
                   bf16* __restrict__ outB, int mode) {
  __shared__ __align__(16) bf16 As[2][TM * LSTR];
  __shared__ __align__(16) bf16 Bs[2][TN * LSTR];

  const int tid  = threadIdx.x;
  const int wave = tid >> 5, lane = tid & 31;
  const int lg   = lane >> 4, ln = lane & 15;
  const int wm   = wave & 3,  wn = wave >> 2;
  const int m0   = blockIdx.x * TM, n0 = blockIdx.y * TN;
  const int ldrow  = tid >> 1;          // 0..127
  const int ldhalf = (tid & 1) * 16;    // 0 or 16 (bf16 elems)

  // LDS byte offsets for async destinations (flat addr low 32 bits = LDS addr)
  const unsigned asb = (unsigned)(size_t)(&As[0][0]);
  const unsigned bsb = (unsigned)(size_t)(&Bs[0][0]);
  const unsigned stg = (unsigned)((ldrow * LSTR + ldhalf) * 2);

  v8f acc[2][4] = {};

  // Stage one K-tile: each thread async-copies 32B of A and 32B of Wt.
  auto stage = [&](int buf, int k0) {
    const bf16* gA = A + (size_t)(m0 + ldrow) * DIN + k0 + ldhalf;
    unsigned la = asb + (unsigned)buf * TILE_BYTES + stg;
    async_ld16(la, gA);
    async_ld16(la + 16, gA + 8);
    const bf16* gB = Wt + (size_t)(n0 + ldrow) * DIN + k0 + ldhalf;
    unsigned lb = bsb + (unsigned)buf * TILE_BYTES + stg;
    async_ld16(lb, gB);
    async_ld16(lb + 16, gB + 8);
  };

  stage(0, 0);

  for (int t = 0; t < NT; ++t) {
    const int cur = t & 1;
    if (t + 1 < NT) {
      stage(cur ^ 1, (t + 1) * TK);
      // 4 newest async loads (next tile) may stay in flight; the 4 oldest
      // (this tile) must have retired (ASYNCcnt completes in order).
      asm volatile("s_wait_asynccnt 0x4" ::: "memory");
    } else {
      asm volatile("s_wait_asynccnt 0x0" ::: "memory");
    }
    __syncthreads();

    const bf16* Asb = &As[cur][0];
    const bf16* Bsb = &Bs[cur][0];

    // A fragment: row = lane&15, K chunks at 8*lg and 16+8*lg.
    v16bf af[2], wf[4];
#pragma unroll
    for (int ms = 0; ms < 2; ++ms) {
      int row = wm * 32 + ms * 16 + ln;
      af[ms] = pack2(*(const v8bf*)(&Asb[row * LSTR + 8 * lg]),
                     *(const v8bf*)(&Asb[row * LSTR + 16 + 8 * lg]));
    }
    // B fragment: col = lane&15, K chunks at 8*lg and 16+8*lg.
#pragma unroll
    for (int ns = 0; ns < 4; ++ns) {
      int col = wn * 64 + ns * 16 + ln;
      wf[ns] = pack2(*(const v8bf*)(&Bsb[col * LSTR + 8 * lg]),
                     *(const v8bf*)(&Bsb[col * LSTR + 16 + 8 * lg]));
    }
#pragma unroll
    for (int ms = 0; ms < 2; ++ms)
#pragma unroll
      for (int ns = 0; ns < 4; ++ns)
        acc[ms][ns] = wmma_bf16(af[ms], wf[ns], acc[ms][ns]);
    __syncthreads();  // all waves done reading buf 'cur' before overwrite
  }

  // Epilogue: C element r -> row m = r + 8*lg, col n = lane&15.
#pragma unroll
  for (int ms = 0; ms < 2; ++ms) {
#pragma unroll
    for (int ns = 0; ns < 4; ++ns) {
      int n = n0 + wn * 64 + ns * 16 + ln;
      float bv = bias[n];
#pragma unroll
      for (int r = 0; r < 8; ++r) {
        int m = m0 + wm * 32 + ms * 16 + r + 8 * lg;
        float vv = acc[ms][ns][r] + bv;
        if (mode == 0) {
          outF[(size_t)m * HID_ + n] = vv;
        } else {
          int bi = m >> 11;          // m / S_
          int si = m & (S_ - 1);
          int hh = n >> 6;           // n / DH_
          int dh = n & (DH_ - 1);
          if (mode == 1)
            outB[((((size_t)bi * H_ + hh) * S_) + si) * DH_ + dh] = (bf16)vv;
          else
            outB[((((size_t)bi * H_ + hh) * DH_) + dh) * S_ + si] = (bf16)vv;
        }
      }
    }
  }
}

// ---- Flash attention with ALiBi, per-wave 16-row query tiles --------------
// Q,K: bf16 [B][H][S][DH]; Vt: bf16 [B][H][DH][S]; O: bf16 [B][S][HID]
// Block = 8 waves -> 128 query rows. Grid: (S/128, H, B).

__global__ __launch_bounds__(256)
void mha_attention(const bf16* __restrict__ Q, const bf16* __restrict__ K,
                   const bf16* __restrict__ Vt, const int* __restrict__ mask,
                   bf16* __restrict__ O) {
  __shared__ __align__(16) bf16 plds[8 * 16 * 32];  // per-wave P staging (1KB)

  const int tid  = threadIdx.x;
  const int wave = tid >> 5, lane = tid & 31;
  const int lg   = lane >> 4, ln = lane & 15;
  const int head = blockIdx.y, bb = blockIdx.z;
  const int qb   = blockIdx.x * 128 + wave * 16;

  const bf16* Qp = Q  + ((size_t)(bb * H_ + head) * S_) * DH_;
  const bf16* Kp = K  + ((size_t)(bb * H_ + head) * S_) * DH_;
  const bf16* Vp = Vt + ((size_t)(bb * H_ + head) * DH_) * S_;
  const int* mrow = mask + bb * S_;
  bf16* pw = &plds[wave * 16 * 32];

  // Q A-fragments for DH chunks [0,32) and [32,64).
  const int qrow = qb + ln;
  v16bf aq0 = pack2(*(const v8bf*)(Qp + (size_t)qrow * DH_ + 8 * lg),
                    *(const v8bf*)(Qp + (size_t)qrow * DH_ + 16 + 8 * lg));
  v16bf aq1 = pack2(*(const v8bf*)(Qp + (size_t)qrow * DH_ + 32 + 8 * lg),
                    *(const v8bf*)(Qp + (size_t)qrow * DH_ + 48 + 8 * lg));

  const float slope = exp2f(-0.5f * (float)(head + 1));  // 2^{-(h+1)*8/H}
  float mrun[8], lrun[8];
  v8f acc[4] = {};
#pragma unroll
  for (int r = 0; r < 8; ++r) { mrun[r] = -1e30f; lrun[r] = 0.f; }

  for (int kb = 0; kb < S_; kb += 32) {
    // ---- scores for two 16-key subtiles --------------------------------
    v8f sf[2];
#pragma unroll
    for (int t = 0; t < 2; ++t) {
      int key = kb + t * 16 + ln;  // this lane's key column
      v16bf bk0 = pack2(*(const v8bf*)(Kp + (size_t)key * DH_ + 8 * lg),
                        *(const v8bf*)(Kp + (size_t)key * DH_ + 16 + 8 * lg));
      v16bf bk1 = pack2(*(const v8bf*)(Kp + (size_t)key * DH_ + 32 + 8 * lg),
                        *(const v8bf*)(Kp + (size_t)key * DH_ + 48 + 8 * lg));
      v8f c = {};
      c = wmma_bf16(aq0, bk0, c);
      c = wmma_bf16(aq1, bk1, c);
      int mk = mrow[key];
#pragma unroll
      for (int r = 0; r < 8; ++r) {
        int qi = qb + r + 8 * lg;
        float s = (c[r] - slope * fabsf((float)(qi - key))) * 0.125f;
        c[r] = (mk != 0) ? s : -1e30f;
      }
      sf[t] = c;
    }

    // ---- online softmax (rows live across 16-lane halves) ---------------
    float corr[8];
#pragma unroll
    for (int r = 0; r < 8; ++r) {
      float v = fmaxf(sf[0][r], sf[1][r]);
      v = fmaxf(v, __shfl_xor(v, 1));
      v = fmaxf(v, __shfl_xor(v, 2));
      v = fmaxf(v, __shfl_xor(v, 4));
      v = fmaxf(v, __shfl_xor(v, 8));
      float mnew = fmaxf(mrun[r], v);
      corr[r] = __expf(mrun[r] - mnew);
      mrun[r] = mnew;
      float p0 = __expf(sf[0][r] - mnew);
      float p1 = __expf(sf[1][r] - mnew);
      pw[(r + 8 * lg) * 32 + ln]      = (bf16)p0;  // stage P in A layout order
      pw[(r + 8 * lg) * 32 + 16 + ln] = (bf16)p1;
      float ls = p0 + p1;
      ls += __shfl_xor(ls, 1);
      ls += __shfl_xor(ls, 2);
      ls += __shfl_xor(ls, 4);
      ls += __shfl_xor(ls, 8);
      lrun[r] = lrun[r] * corr[r] + ls;
    }
#pragma unroll
    for (int n = 0; n < 4; ++n)
#pragma unroll
      for (int r = 0; r < 8; ++r)
        acc[n][r] *= corr[r];

    // ---- P * V (K = 32 keys), V read from transposed layout -------------
    v16bf ap = pack2(*(const v8bf*)(&pw[ln * 32 + 8 * lg]),
                     *(const v8bf*)(&pw[ln * 32 + 16 + 8 * lg]));
#pragma unroll
    for (int n = 0; n < 4; ++n) {
      int dh = n * 16 + ln;
      v16bf bv = pack2(*(const v8bf*)(Vp + (size_t)dh * S_ + kb + 8 * lg),
                       *(const v8bf*)(Vp + (size_t)dh * S_ + kb + 16 + 8 * lg));
      acc[n] = wmma_bf16(ap, bv, acc[n]);
    }
  }

  // ---- normalize and store [B,S,HID] bf16 -------------------------------
#pragma unroll
  for (int n = 0; n < 4; ++n) {
#pragma unroll
    for (int r = 0; r < 8; ++r) {
      int qi  = qb + r + 8 * lg;
      int col = head * DH_ + n * 16 + ln;
      float o = acc[n][r] / lrun[r];
      O[((size_t)bb * S_ + qi) * HID_ + col] = (bf16)o;
    }
  }
}

// ---------------------------------------------------------------------------

extern "C" void kernel_launch(void* const* d_in, const int* in_sizes, int n_in,
                              void* d_out, int out_size, void* d_ws, size_t ws_size,
                              hipStream_t stream) {
  (void)in_sizes; (void)n_in; (void)out_size; (void)ws_size;
  const float* x    = (const float*)d_in[0];
  const float* Wq   = (const float*)d_in[1];
  const float* bq   = (const float*)d_in[2];
  const float* Wk   = (const float*)d_in[3];
  const float* bk   = (const float*)d_in[4];
  const float* Wv   = (const float*)d_in[5];
  const float* bv   = (const float*)d_in[6];
  const float* Wo   = (const float*)d_in[7];
  const float* bo   = (const float*)d_in[8];
  const int*   pmsk = (const int*)d_in[9];
  float* out = (float*)d_out;

  // Workspace layout (bf16 elements): ~92 MB total.
  bf16* w   = (bf16*)d_ws;
  bf16* xb  = w;  w += (size_t)M_ * DIN;     // x in bf16
  bf16* wqT = w;  w += (size_t)DIN * HID_;   // W^T bf16
  bf16* wkT = w;  w += (size_t)DIN * HID_;
  bf16* wvT = w;  w += (size_t)DIN * HID_;
  bf16* woT = w;  w += (size_t)DIN * HID_;
  bf16* qB  = w;  w += (size_t)M_ * HID_;    // [B,H,S,DH]
  bf16* kB  = w;  w += (size_t)M_ * HID_;    // [B,H,S,DH]
  bf16* vB  = w;  w += (size_t)M_ * HID_;    // [B,H,DH,S]
  bf16* aB  = w;  w += (size_t)M_ * HID_;    // attention out [B,S,HID]

  const int nx = M_ * DIN;
  mha_cvt_bf16<<<(nx + 255) / 256, 256, 0, stream>>>(x, xb, nx);
  const int nw = DIN * HID_;
  mha_cvt_bf16_T<<<(nw + 255) / 256, 256, 0, stream>>>(Wq, wqT, DIN, HID_);
  mha_cvt_bf16_T<<<(nw + 255) / 256, 256, 0, stream>>>(Wk, wkT, DIN, HID_);
  mha_cvt_bf16_T<<<(nw + 255) / 256, 256, 0, stream>>>(Wv, wvT, DIN, HID_);
  mha_cvt_bf16_T<<<(nw + 255) / 256, 256, 0, stream>>>(Wo, woT, DIN, HID_);

  dim3 gg(M_ / TM, HID_ / TN);  // 64 x 8 blocks
  mha_gemm_bf16<<<gg, 256, 0, stream>>>(xb, wqT, bq, nullptr, qB, 1);
  mha_gemm_bf16<<<gg, 256, 0, stream>>>(xb, wkT, bk, nullptr, kB, 1);
  mha_gemm_bf16<<<gg, 256, 0, stream>>>(xb, wvT, bv, nullptr, vB, 2);

  mha_attention<<<dim3(S_ / 128, H_, B_), 256, 0, stream>>>(qB, kB, vB, pmsk, aB);

  mha_gemm_bf16<<<gg, 256, 0, stream>>>(aB, woT, bo, out, nullptr, 0);
}
